// FlowAttention_35330400977338
// MI455X (gfx1250) — compile-verified, matrix operand order checked by reading, not run on hardware
//
#include <hip/hip_runtime.h>
#include <hip/hip_bf16.h>

#define BB 16
#define NN 4096
#define DD 256
#define EPS 1e-6f

typedef __attribute__((ext_vector_type(16))) __bf16 v16bf;
typedef __attribute__((ext_vector_type(8)))  __bf16 v8bf;
typedef __attribute__((ext_vector_type(8)))  float  v8f;

// Fast sigmoid: v_rcp_f32 instead of IEEE divide sequence.
__device__ __forceinline__ float sigmoidf(float x) {
    return __builtin_amdgcn_rcpf(1.0f + __expf(-x));
}

// Load one 16x32 bf16 WMMA operand half-row for this lane.
// WMMA 16-bit A/B layout: VGPR0-3 hold K = half*8 + 0..7 (contiguous),
// VGPR4-7 hold K = half*8 + 16..23.  `p` points at (row_base + half*8).
__device__ __forceinline__ v16bf load_op(const __bf16* __restrict__ p) {
    v8bf lo = *(const v8bf*)p;          // K offsets 0..7
    v8bf hi = *(const v8bf*)(p + 16);   // K offsets 16..23
    return __builtin_shufflevector(lo, hi, 0, 1, 2, 3, 4, 5, 6, 7,
                                   8, 9, 10, 11, 12, 13, 14, 15);
}

// ---------------------------------------------------------------------------
// f32 -> bf16, 8 elements/thread (2x b128 load, 1x b128 store).  n % 8 == 0.
// ---------------------------------------------------------------------------
__global__ void conv_bf16_kernel(const float* __restrict__ src,
                                 __bf16* __restrict__ dst, long n)
{
    long i = ((long)blockIdx.x * blockDim.x + threadIdx.x) * 8;
    if (i >= n) return;
    v8bf o;
#pragma unroll
    for (int j = 0; j < 8; ++j) o[j] = (__bf16)src[i + j];
    *(v8bf*)(dst + i) = o;
}

// f32 [D,D] -> bf16 transposed [D,D] (dst[c][r] = src[r][c]).
__global__ void conv_wt_kernel(const float* __restrict__ src,
                               __bf16* __restrict__ dst)
{
    int t = blockIdx.x * blockDim.x + threadIdx.x;   // 0 .. D*D-1
    int r = t >> 8, c = t & 255;
    dst[(size_t)c * DD + r] = (__bf16)src[(size_t)r * DD + c];
}

// ---------------------------------------------------------------------------
// phi = sigmoid(X @ W + bias).  Block: 16 rows x 256 cols; 8 waves, 2 tiles/wave.
// X is bf16 row-major [B*N, D]; WT is bf16 transposed (WT[e][d] = W[d][e]).
// mode 0 (Q): write I = phi/(rowsum+EPS) bf16 row-major, gate = sigmoid(rowsum)
// mode 1 (K): write phi bf16 TRANSPOSED [B, D, N], atomic column sums
// ---------------------------------------------------------------------------
__global__ void proj_kernel(const __bf16* __restrict__ Xbf,
                            const __bf16* __restrict__ WT,
                            const float* __restrict__ bias,
                            __bf16* __restrict__ phiOut,
                            float* __restrict__ gate,
                            float* __restrict__ colsum,
                            int mode)
{
    __shared__ float rowsumS[16];
    const int lane = threadIdx.x & 31;
    const int wv   = threadIdx.x >> 5;
    const int half = lane >> 4;
    const int ln16 = lane & 15;
    const int rowb = blockIdx.x * 16;        // global row in [0, B*N)
    const int bidx = rowb >> 12;             // row / N
    const int nloc = rowb & (NN - 1);        // row % N

    if (threadIdx.x < 16) rowsumS[threadIdx.x] = 0.0f;
    __syncthreads();

    const int cb0 = wv * 32;
    const int cb1 = cb0 + 16;
    const __bf16* aRow  = Xbf + (size_t)(rowb + ln16) * DD + half * 8;
    const __bf16* bRow0 = WT  + (size_t)(cb0 + ln16) * DD + half * 8;
    const __bf16* bRow1 = WT  + (size_t)(cb1 + ln16) * DD + half * 8;

    v8f acc0 = {}; v8f acc1 = {};
    for (int kb = 0; kb < DD; kb += 32) {
        v16bf a  = load_op(aRow + kb);
        v16bf b0 = load_op(bRow0 + kb);
        v16bf b1 = load_op(bRow1 + kb);
        acc0 = __builtin_amdgcn_wmma_f32_16x16x32_bf16(false, a, false, b0,
                                                       (short)0, acc0, false, false);
        acc1 = __builtin_amdgcn_wmma_f32_16x16x32_bf16(false, a, false, b1,
                                                       (short)0, acc1, false, false);
    }

    float phi0[8], phi1[8];
    float cs0 = 0.0f, cs1 = 0.0f;
    const float bia0 = bias[cb0 + ln16];
    const float bia1 = bias[cb1 + ln16];
#pragma unroll
    for (int r = 0; r < 8; ++r) {
        int m = (half << 3) + r;
        phi0[r] = sigmoidf(acc0[r] + bia0);
        phi1[r] = sigmoidf(acc1[r] + bia1);
        if (mode == 0) {
            atomicAdd(&rowsumS[m], phi0[r] + phi1[r]);   // per-row feature sum
        } else {
            cs0 += phi0[r];                               // per-column partials
            cs1 += phi1[r];
        }
    }

    if (mode == 0) {
        __syncthreads();
#pragma unroll
        for (int r = 0; r < 8; ++r) {
            int m = (half << 3) + r;
            float irs = __builtin_amdgcn_rcpf(rowsumS[m] + EPS);
            size_t row = (size_t)(rowb + m);
            phiOut[row * DD + cb0 + ln16] = (__bf16)(phi0[r] * irs);
            phiOut[row * DD + cb1 + ln16] = (__bf16)(phi1[r] * irs);
        }
        if (threadIdx.x < 16)
            gate[rowb + threadIdx.x] = sigmoidf(rowsumS[threadIdx.x]);
    } else {
        // transposed store: phiOut[b][d][n]
        __bf16* base = phiOut + (size_t)bidx * DD * NN;
#pragma unroll
        for (int r = 0; r < 8; ++r) {
            int m = (half << 3) + r;
            base[(size_t)(cb0 + ln16) * NN + nloc + m] = (__bf16)phi0[r];
            base[(size_t)(cb1 + ln16) * NN + nloc + m] = (__bf16)phi1[r];
        }
        atomicAdd(&colsum[bidx * DD + cb0 + ln16], cs0);
        atomicAdd(&colsum[bidx * DD + cb1 + ln16], cs1);
    }
}

// ---------------------------------------------------------------------------
// V_b = softmax over sequence of (phi_k/(colsum+EPS)) * V, per (b,d) column.
// phiK and Vb are transposed [B,D,N] -> fully coalesced scans over n.
// ---------------------------------------------------------------------------
__global__ void softmax_col_kernel(const __bf16* __restrict__ phiKT,
                                   const float* __restrict__ V,
                                   const float* __restrict__ colsum,
                                   __bf16* __restrict__ VbT)
{
    __shared__ float red[256];
    const int b = blockIdx.x / DD;
    const int d = blockIdx.x % DD;
    const float ics = __builtin_amdgcn_rcpf(colsum[b * DD + d] + EPS);
    const __bf16* pk = phiKT + ((size_t)b * DD + d) * NN;
    __bf16* vb = VbT + ((size_t)b * DD + d) * NN;

    float vals[16];
#pragma unroll
    for (int i = 0; i < 16; ++i) {
        int n = threadIdx.x + i * 256;
        vals[i] = ((float)pk[n] * ics) * V[((size_t)b * NN + n) * DD + d];
    }
    float mx = vals[0];
#pragma unroll
    for (int i = 1; i < 16; ++i) mx = fmaxf(mx, vals[i]);
    red[threadIdx.x] = mx;
    __syncthreads();
    for (int s = 128; s > 0; s >>= 1) {
        if (threadIdx.x < s)
            red[threadIdx.x] = fmaxf(red[threadIdx.x], red[threadIdx.x + s]);
        __syncthreads();
    }
    mx = red[0];
    __syncthreads();

    float se = 0.0f;
#pragma unroll
    for (int i = 0; i < 16; ++i) { vals[i] = __expf(vals[i] - mx); se += vals[i]; }
    red[threadIdx.x] = se;
    __syncthreads();
    for (int s = 128; s > 0; s >>= 1) {
        if (threadIdx.x < s)
            red[threadIdx.x] += red[threadIdx.x + s];
        __syncthreads();
    }
    float inv = __builtin_amdgcn_rcpf(red[0]);
#pragma unroll
    for (int i = 0; i < 16; ++i)
        vb[threadIdx.x + i * 256] = (__bf16)(vals[i] * inv);
}

// ---------------------------------------------------------------------------
// KV[b] = phi_k[b]^T @ V_b[b].  Both operands stored [B,D,N]: the lane's row
// is a d-index and K (=n) is contiguous -> pure b128 operand loads.
// Output stored transposed: KVT[b][e][d].
// ---------------------------------------------------------------------------
__global__ void kv_kernel(const __bf16* __restrict__ phiKT,
                          const __bf16* __restrict__ VbT,
                          __bf16* __restrict__ KVT)
{
    const int lane = threadIdx.x & 31;
    const int half = lane >> 4;
    const int ln16 = lane & 15;
    const int b  = blockIdx.x >> 8;
    const int t  = blockIdx.x & 255;
    const int dm = (t >> 4) << 4;
    const int dn = (t & 15) << 4;
    const __bf16* aRow = phiKT + ((size_t)b * DD + dm + ln16) * NN + half * 8;
    const __bf16* bRow = VbT   + ((size_t)b * DD + dn + ln16) * NN + half * 8;

    v8f acc = {};
    for (int n0 = 0; n0 < NN; n0 += 32) {
        v16bf a  = load_op(aRow + n0);
        v16bf bm = load_op(bRow + n0);
        acc = __builtin_amdgcn_wmma_f32_16x16x32_bf16(false, a, false, bm,
                                                      (short)0, acc, false, false);
    }
    __bf16* o = KVT + (size_t)b * DD * DD;
#pragma unroll
    for (int r = 0; r < 8; ++r) {
        int m = (half << 3) + r;                    // d-row index
        o[(size_t)(dn + ln16) * DD + dm + m] = (__bf16)acc[r];   // KVT[e][d]
    }
}

// ---------------------------------------------------------------------------
// R = gate * (I @ KV).  I row-major [B*N, D]; KVT[e][d] makes B K-contiguous.
// ---------------------------------------------------------------------------
__global__ void out_kernel(const __bf16* __restrict__ Ibf,
                           const __bf16* __restrict__ KVT,
                           const float* __restrict__ gate,
                           float* __restrict__ out)
{
    const int lane = threadIdx.x & 31;
    const int wv   = threadIdx.x >> 5;
    const int half = lane >> 4;
    const int ln16 = lane & 15;
    const int rowb = blockIdx.x * 16;
    const int b    = rowb >> 12;
    const int cb0 = wv * 32;
    const int cb1 = cb0 + 16;
    const __bf16* aRow  = Ibf + (size_t)(rowb + ln16) * DD + half * 8;
    const __bf16* bRow0 = KVT + ((size_t)b * DD + cb0 + ln16) * DD + half * 8;
    const __bf16* bRow1 = KVT + ((size_t)b * DD + cb1 + ln16) * DD + half * 8;

    v8f acc0 = {}; v8f acc1 = {};
    for (int k0 = 0; k0 < DD; k0 += 32) {
        v16bf a  = load_op(aRow + k0);
        v16bf b0 = load_op(bRow0 + k0);
        v16bf b1 = load_op(bRow1 + k0);
        acc0 = __builtin_amdgcn_wmma_f32_16x16x32_bf16(false, a, false, b0,
                                                       (short)0, acc0, false, false);
        acc1 = __builtin_amdgcn_wmma_f32_16x16x32_bf16(false, a, false, b1,
                                                       (short)0, acc1, false, false);
    }
#pragma unroll
    for (int r = 0; r < 8; ++r) {
        int m = (half << 3) + r;
        float g = gate[rowb + m];
        out[(size_t)(rowb + m) * DD + cb0 + ln16] = g * acc0[r];
        out[(size_t)(rowb + m) * DD + cb1 + ln16] = g * acc1[r];
    }
}

// ---------------------------------------------------------------------------
extern "C" void kernel_launch(void* const* d_in, const int* in_sizes, int n_in,
                              void* d_out, int out_size, void* d_ws, size_t ws_size,
                              hipStream_t stream)
{
    const float* Q  = (const float*)d_in[0];
    const float* K  = (const float*)d_in[1];
    const float* V  = (const float*)d_in[2];
    const float* Wq = (const float*)d_in[3];
    const float* bq = (const float*)d_in[4];
    const float* Wk = (const float*)d_in[5];
    const float* bk = (const float*)d_in[6];
    float* out = (float*)d_out;

    // Workspace layout (~128.4 MB).  Buffer reuse is safe by stream order:
    //   Qbf dead after proj(Q)  -> reused for KVT (kv_kernel runs later)
    //   Kbf dead after proj(K)  -> reused for VbT (softmax runs later)
    char* ws = (char*)d_ws;
    const size_t bnD2 = (size_t)BB * NN * DD * 2;   // 32 MB (bf16 [B,N,D])
    __bf16* Qbf   = (__bf16*)(ws);                  // later: KVT (2 MB)
    __bf16* Kbf   = (__bf16*)(ws + bnD2);           // later: VbT (32 MB)
    __bf16* phiKT = (__bf16*)(ws + 2 * bnD2);       // [B,D,N]
    __bf16* Ibf   = (__bf16*)(ws + 3 * bnD2);       // [B*N,D]
    __bf16* WqT   = (__bf16*)(ws + 4 * bnD2);                        // 128 KB
    __bf16* WkT   = (__bf16*)(ws + 4 * bnD2 + (size_t)DD * DD * 2);  // 128 KB
    float* colsum = (float*)(ws + 4 * bnD2 + 2 * (size_t)DD * DD * 2);
    float* gate   = (float*)(ws + 4 * bnD2 + 2 * (size_t)DD * DD * 2
                                + (size_t)BB * DD * 4);
    __bf16* KVT = Qbf;   // alias (Qbf dead before kv_kernel)
    __bf16* VbT = Kbf;   // alias (Kbf dead before softmax)

    hipMemsetAsync(colsum, 0, (size_t)BB * DD * sizeof(float), stream);

    const long nElems = (long)BB * NN * DD;
    const int convBlocks = (int)(nElems / 8 / 256);     // 8192
    conv_bf16_kernel<<<convBlocks, 256, 0, stream>>>(Q, Qbf, nElems);
    conv_bf16_kernel<<<convBlocks, 256, 0, stream>>>(K, Kbf, nElems);
    conv_wt_kernel<<<DD * DD / 256, 256, 0, stream>>>(Wq, WqT);
    conv_wt_kernel<<<DD * DD / 256, 256, 0, stream>>>(Wk, WkT);

    const int rowTiles = BB * NN / 16;   // 4096 blocks of 16 rows
    dim3 blk(256);
    proj_kernel<<<rowTiles, blk, 0, stream>>>(Kbf, WkT, bk, phiKT, nullptr, colsum, 1);
    proj_kernel<<<rowTiles, blk, 0, stream>>>(Qbf, WqT, bq, Ibf, gate, nullptr, 0);
    softmax_col_kernel<<<BB * DD, blk, 0, stream>>>(phiKT, V, colsum, VbT);
    kv_kernel<<<BB * 256, dim3(32), 0, stream>>>(phiKT, VbT, KVT);
    out_kernel<<<rowTiles, blk, 0, stream>>>(Ibf, KVT, gate, out);
}
